// ResidualVQ_81398220194538
// MI455X (gfx1250) — compile-verified, matrix operand order checked by reading, not run on hardware
//
#include <hip/hip_runtime.h>
#include <hip/hip_bf16.h>

// ---------------------------------------------------------------------------
// ResidualVQ forward for MI455X (gfx1250).
// bf16 WMMA distance GEMM + TDM (tensor_load_to_lds) staging of codebook
// tiles into LDS, shared by all 4 waves of the workgroup.
// Sizes (fixed by reference): B=8, D=256, T=4096, Q=8, K=1024.
// d_out layout (flat fp32): [quantized 8*256*4096][loss 1][indices 8*8*4096]
// d_ws  layout: [codebook bf16: 8*1024*256 ushort = 4 MiB][enorm: 8192 f32]
// ---------------------------------------------------------------------------

#define QSTAGES 8
#define KCODES  1024
#define DDIM    256
#define TLEN    4096
#define BBATCH  8
#define ROWS    64          // tokens per block
#define RS      257         // LDS row stride (floats), conflict-free
#define NQUANT  (BBATCH * DDIM * TLEN)       // 8388608
#define LOSS_OFF ((size_t)NQUANT)
#define IDX_OFF  ((size_t)NQUANT + 1)

#define SUPER_CODES 64      // codes per TDM super-tile (32 KiB bf16)
#define NSUPER      (KCODES / SUPER_CODES)   // 16

typedef __attribute__((ext_vector_type(16))) __bf16 v16bf;
typedef __attribute__((ext_vector_type(8)))  float  v8f;
typedef unsigned int u32x4 __attribute__((ext_vector_type(4)));
typedef int          i32x8 __attribute__((ext_vector_type(8)));
typedef int          i32x4 __attribute__((ext_vector_type(4)));

union FragU {
    v16bf          bf;
    uint4          q4[2];
    unsigned short us[16];
};

__device__ __forceinline__ unsigned short f2bf(float f) {
    // round-to-nearest-even fp32 -> bf16
    unsigned u = __float_as_uint(f);
    unsigned r = u + 0x7FFFu + ((u >> 16) & 1u);
    return (unsigned short)(r >> 16);
}

// 1-D TDM transfer: 4096 elements of 8 bytes (32 KiB) global -> LDS.
// D# group0: count=1 | lds_addr | global_addr[56:0] | type=2
// D# group1: data_size=3 (8B); tensor_dim0=1<<20; tensor_dim1=1;
//            tile_dim0=4096; tile_dim1=1; tensor_dim0_stride=4096.
__device__ __forceinline__ void tdm_load_32k(unsigned int lds_off,
                                             const void* gsrc) {
    unsigned long long ga = (unsigned long long)(uintptr_t)gsrc;
    u32x4 g0;
    g0.x = 1u;                                   // count=1, no gather
    g0.y = lds_off;                              // LDS byte address
    g0.z = (unsigned int)ga;                     // global_addr[31:0]
    g0.w = (unsigned int)(ga >> 32) | (2u << 30);// global_addr[56:32] | type=2
    i32x8 g1 = { (int)0x00030000,   // data_size=3 (8B), mask=0
                 0,                 // atomic barrier addr = 0
                 (int)0x00010010,   // tensor_dim0=1<<20 (hi), tensor_dim1=1 (lo)
                 (int)0x10000000,   // tile_dim0=4096 in [31:16]
                 1,                 // tile_dim1=1
                 4096,              // tensor_dim0_stride[31:0]
                 0, 0 };
    i32x4 zz = {0, 0, 0, 0};
    i32x8 z8 = {0, 0, 0, 0, 0, 0, 0, 0};
    // 6-arg form (clang-23 / therock-10.0 headers)
    __builtin_amdgcn_tensor_load_to_lds(g0, g1, zz, zz, z8, 0);
}

// ---------------------------------------------------------------------------
// Prep: convert codebooks fp32 -> bf16 into ws, zero the loss accumulator.
// ---------------------------------------------------------------------------
__global__ void rvq_prep_convert(const float* __restrict__ cb,
                                 unsigned short* __restrict__ cb_bf,
                                 float* __restrict__ out) {
    const size_t n = (size_t)QSTAGES * KCODES * DDIM;   // 2,097,152
    for (size_t i = blockIdx.x * (size_t)blockDim.x + threadIdx.x; i < n;
         i += (size_t)gridDim.x * blockDim.x)
        cb_bf[i] = f2bf(cb[i]);
    if (blockIdx.x == 0 && threadIdx.x == 0) out[LOSS_OFF] = 0.0f;
}

// Prep: per-code squared norms ||e_k||^2 (fp32).
__global__ void rvq_prep_enorm(const float* __restrict__ cb,
                               float* __restrict__ enorm) {
    int c = blockIdx.x * blockDim.x + threadIdx.x;      // 0..8191
    if (c < QSTAGES * KCODES) {
        const float* p = cb + (size_t)c * DDIM;
        float s = 0.0f;
        #pragma unroll 8
        for (int d = 0; d < DDIM; ++d) { float v = p[d]; s += v * v; }
        enorm[c] = s;
    }
}

// ---------------------------------------------------------------------------
// Main: each block owns 64 consecutive tokens; 4 waves x 16 rows each.
// ---------------------------------------------------------------------------
__global__ __launch_bounds__(128) void rvq_main(
        const float* __restrict__ z,
        const float* __restrict__ cb,               // fp32 codebooks [Q,K,D]
        const unsigned short* __restrict__ cb_bf,   // bf16 codebooks [Q,K,D]
        const float* __restrict__ enorm,            // [Q,K]
        float* __restrict__ out) {

    __shared__ float res[ROWS * RS];     // fp32 residual tile [64][256] padded
    __shared__ unsigned short Bbuf[2][SUPER_CODES * DDIM];  // 2 x 32 KiB
    __shared__ int   idx_s[ROWS];
    __shared__ float wloss[4];

    const int tid  = threadIdx.x;
    const int lane = tid & 31;
    const int wave = tid >> 5;
    const int tok0 = blockIdx.x * ROWS;         // 64 | 4096 -> single batch b
    const int b    = tok0 >> 12;
    const int t0   = tok0 & (TLEN - 1);

    const int col = lane & 15;   // code column in tile; also row id for A frag
    const int hi  = lane >> 4;   // K-half selector per WMMA 16-bit layout

    const unsigned int ldsB0 =
        (unsigned int)(uintptr_t)&Bbuf[0][0];   // LDS byte offset of buffer 0
    const unsigned int ldsB1 =
        (unsigned int)(uintptr_t)&Bbuf[1][0];

    // --- load z tile transposed into LDS: res[tok][d] = z[b][d][t0+tok] ---
    for (int e = tid; e < ROWS * DDIM; e += 128) {
        int tokl = e & (ROWS - 1);
        int d    = e >> 6;
        res[tokl * RS + d] = z[(((size_t)(b * DDIM + d)) << 12) + t0 + tokl];
    }
    float lossAcc = 0.0f;
    __syncthreads();

    for (int q = 0; q < QSTAGES; ++q) {
        const unsigned short* cbq = cb_bf + (size_t)q * KCODES * DDIM;
        const float*          enq = enorm + q * KCODES;

        // prologue: TDM super-tile 0 into buffer 0 (wave 0 only; overlaps
        // with A-fragment construction below)
        if (wave == 0) tdm_load_32k(ldsB0, cbq);

        // ---- build A fragments (16 rows of this wave), bf16 ----
        FragU A[8];
        {
            const int row = wave * 16 + col;
            const float* rp = &res[row * RS];
            #pragma unroll
            for (int c = 0; c < 8; ++c) {
                const int base = c * 32 + (hi ? 8 : 0);
                #pragma unroll
                for (int j = 0; j < 8; ++j) A[c].us[j]     = f2bf(rp[base + j]);
                #pragma unroll
                for (int j = 0; j < 8; ++j) A[c].us[8 + j] = f2bf(rp[base + 16 + j]);
            }
        }

        float minv[8];
        int   mini[8];
        #pragma unroll
        for (int j = 0; j < 8; ++j) { minv[j] = 3.4e38f; mini[j] = 0; }

        // ---- 16 super-tiles of 64 codes; double-buffered TDM pipeline ----
        for (int sp = 0; sp < NSUPER; ++sp) {
            const int cur = sp & 1;
            if (wave == 0) {
                if (sp + 1 < NSUPER) {
                    tdm_load_32k(cur ? ldsB0 : ldsB1,
                                 cbq + (size_t)(sp + 1) * SUPER_CODES * DDIM);
                    __builtin_amdgcn_s_wait_tensorcnt(1); // super-tile sp done
                } else {
                    __builtin_amdgcn_s_wait_tensorcnt(0); // last one done
                }
            }
            __syncthreads();      // LDS super-tile `sp` visible to all waves

            const unsigned short* sb = &Bbuf[cur][0];
            #pragma unroll
            for (int tt = 0; tt < 4; ++tt) {
                const int code = sp * SUPER_CODES + tt * 16 + col;
                const unsigned short* bp =
                    sb + (tt * 16 + col) * DDIM + (hi ? 8 : 0);

                FragU Bf[8];
                #pragma unroll
                for (int c = 0; c < 8; ++c) {            // ds_load_b128 x16
                    Bf[c].q4[0] = *(const uint4*)(bp + c * 32);
                    Bf[c].q4[1] = *(const uint4*)(bp + c * 32 + 16);
                }
                const float en = enq[code];

                v8f acc = {};
                #pragma unroll
                for (int c = 0; c < 8; ++c)
                    acc = __builtin_amdgcn_wmma_f32_16x16x32_bf16(
                            false, A[c].bf, false, Bf[c].bf,
                            (short)0, acc, false, false);

                #pragma unroll
                for (int j = 0; j < 8; ++j) {
                    float dv = en - 2.0f * acc[j];       // ||e||^2 - 2 r.e
                    if (dv < minv[j]) { minv[j] = dv; mini[j] = code; }
                }
            }
            __syncthreads();      // everyone done reading buf[cur]
        }

        // ---- argmin reduce across the 16 lanes of each half-group ----
        #pragma unroll
        for (int off = 1; off < 16; off <<= 1) {
            #pragma unroll
            for (int j = 0; j < 8; ++j) {
                float ov = __shfl_xor(minv[j], off, 32);
                int   oi = __shfl_xor(mini[j], off, 32);
                if (ov < minv[j] || (ov == minv[j] && oi < mini[j])) {
                    minv[j] = ov; mini[j] = oi;
                }
            }
        }

        // lanes 0 (rows 0..7) and 16 (rows 8..15) publish indices
        if (col == 0) {
            #pragma unroll
            for (int j = 0; j < 8; ++j) {
                int m   = j + 8 * hi;
                int row = wave * 16 + m;
                idx_s[row] = mini[j];
                size_t tok = (size_t)tok0 + row;
                out[IDX_OFF + (size_t)q * (BBATCH * TLEN) + tok] = (float)mini[j];
            }
        }
        __syncthreads();

        // ---- exact fp32 residual update + loss accumulation ----
        for (int e = tid; e < ROWS * DDIM; e += 128) {
            int tokl = e & (ROWS - 1);
            int d    = e >> 6;
            int idx  = idx_s[tokl];
            float rv = res[tokl * RS + d]
                     - cb[((size_t)(q * KCODES + idx)) * DDIM + d];
            res[tokl * RS + d] = rv;
            lossAcc += rv * rv;                 // (z_q - r)^2 == rv^2
        }
        __syncthreads();
    }

    // ---- quantized_out = z - residual_final (store transposed back) ----
    for (int e = tid; e < ROWS * DDIM; e += 128) {
        int tokl = e & (ROWS - 1);
        int d    = e >> 6;
        size_t gi = (((size_t)(b * DDIM + d)) << 12) + t0 + tokl;
        out[gi] = z[gi] - res[tokl * RS + d];
    }

    // ---- loss block-reduce + single atomic per block ----
    #pragma unroll
    for (int off = 16; off >= 1; off >>= 1)
        lossAcc += __shfl_xor(lossAcc, off, 32);
    if (lane == 0) wloss[wave] = lossAcc;
    __syncthreads();
    if (tid == 0) {
        float s = wloss[0] + wloss[1] + wloss[2] + wloss[3];
        // total_loss = sum_q 1.25 * mean(rv^2) ; mean over B*T*D = 8388608
        atomicAdd(out + LOSS_OFF, s * (1.25f / 8388608.0f));
    }
}

// ---------------------------------------------------------------------------
extern "C" void kernel_launch(void* const* d_in, const int* in_sizes, int n_in,
                              void* d_out, int out_size, void* d_ws, size_t ws_size,
                              hipStream_t stream) {
    const float* z  = (const float*)d_in[0];   // [8,256,4096] fp32
    const float* cb = (const float*)d_in[1];   // [8,1024,256] fp32
    float* out = (float*)d_out;

    unsigned short* cb_bf = (unsigned short*)d_ws;                       // 4 MiB
    float* enorm = (float*)((char*)d_ws + (size_t)QSTAGES * KCODES * DDIM * 2);

    rvq_prep_convert<<<256, 256, 0, stream>>>(cb, cb_bf, out);
    rvq_prep_enorm<<<(QSTAGES * KCODES + 255) / 256, 256, 0, stream>>>(cb, enorm);

    // 32768 tokens / 64 per block = 512 blocks, 128 threads (4 waves)
    rvq_main<<<512, 128, 0, stream>>>(z, cb, cb_bf, enorm, out);
}
